// TrajectoryImitationSelfCNNLSTMWithRasterizer_69380901700104
// MI455X (gfx1250) — matrix operand order repaired and checked by reading, not compiled, not run on hardware
//
#include <hip/hip_runtime.h>
#include <math.h>

// ---------------- types for gfx1250 WMMA (wave32) ----------------
typedef __bf16 v16bf __attribute__((ext_vector_type(16)));
typedef float  v8f   __attribute__((ext_vector_type(8)));

struct alignas(16) P128 { unsigned int x, y, z, w; };   // POD 16B block

union Frag {                 // one lane's 8-VGPR WMMA operand
    v16bf v;
    unsigned short h[16];
    P128 q[2];
};

__device__ __forceinline__ v8f vzero() {
    v8f z = {0.f, 0.f, 0.f, 0.f, 0.f, 0.f, 0.f, 0.f};
    return z;
}
__device__ __forceinline__ v8f wmma_bf16(const Frag& a, const Frag& b, v8f c) {
    // D(16x16,f32) = A(16x32,bf16) * B(32x16,bf16) + C
    return __builtin_amdgcn_wmma_f32_16x16x32_bf16(false, a.v, false, b.v,
                                                   (short)0, c, false, false);
}
__device__ __forceinline__ unsigned short f2bf(float x) {   // RNE f32->bf16
    union { float f; unsigned int u; } c; c.f = x;
    unsigned int r = c.u + 0x7FFFu + ((c.u >> 16) & 1u);
    return (unsigned short)(r >> 16);
}
__device__ __forceinline__ float bf2f(unsigned short h) {
    union { float f; unsigned int u; } c; c.u = ((unsigned int)h) << 16;
    return c.f;
}
__device__ __forceinline__ float sigm(float x) { return 1.f / (1.f + __expf(-x)); }

// ---------------- model constants ----------------
#define HGT 200
#define WID 200
// conv tile: 2 conv2-output rows per workgroup
#define C1_ROWS 5            // conv1 rows needed: 2*Y0 .. 2*Y0+4
#define IMG_ROWS 7           // img rows needed: 2*Y0-1 .. 2*Y0+5
#define IMG_W 210            // img cols, shifted +1, zero-padded halo
#define C1_W 208             // conv1 cols 0..199 + zero pad to 208

// =================================================================
// Kernel 1: fused conv1(12->3, 3x3, s1, SAME) + conv2(3->32, 3x3, s2,
//           SAME) + relu + spatial-sum per channel, all via WMMA bf16.
// conv1 K-order = tap*16 + ci (channels padded to 16) so every wave32
// A-chunk is 16 contiguous LDS bytes -> ds_load_b128, conflict-free.
// grid = (64 batches, 50 row-tiles), block = 256 (8 waves)
// =================================================================
__global__ __launch_bounds__(256) void k_conv(
    const float* __restrict__ img, const float* __restrict__ comp_w,
    const float* __restrict__ comp_b, const float* __restrict__ cnn_w,
    const float* __restrict__ cnn_b, float* __restrict__ z_sum)
{
    __shared__ unsigned short imgT[IMG_ROWS * IMG_W * 16]; // bf16, channel-last
    __shared__ unsigned short c1T[3 * C1_ROWS * C1_W];     // bf16 conv1 out
    __shared__ int   k2Base[32];                           // conv2 K -> offset
    __shared__ float accLds[32];

    const int tid  = threadIdx.x;
    const int lane = tid & 31, wave = tid >> 5;
    const int half = lane >> 4, lm = lane & 15;
    const int b  = blockIdx.x;
    const int Y0 = blockIdx.y * 2;          // first conv2 output row of tile

    // ---- phase A: zero LDS (covers halo + channel pad 12..15) ----
    {
        unsigned int* z32 = (unsigned int*)imgT;
        for (int i = tid; i < IMG_ROWS * IMG_W * 8; i += 256) z32[i] = 0;
        for (int i = tid; i < 3 * C1_ROWS * C1_W; i += 256) c1T[i] = 0;
    }
    if (tid < 32) {
        int c1 = tid / 9, tap = tid % 9, u = tap / 3, v = tap % 3;
        k2Base[tid] = (tid < 27) ? (c1 * (C1_ROWS * C1_W) + u * C1_W + v) : 0;
        accLds[tid] = 0.f;
    }
    __syncthreads();

    // ---- phase B: stage image tile channel-last, 2 bf16 per b32 store ----
    {
        unsigned int* imgT32 = (unsigned int*)imgT;
        for (int e = tid; e < 6 * IMG_ROWS * 200; e += 256) {
            int ci2 = e / (IMG_ROWS * 200);            // channel pair 0..5
            int rem = e - ci2 * (IMG_ROWS * 200);
            int yi = rem / 200, x = rem % 200;
            int gy = 2 * Y0 - 1 + yi;
            if (gy >= 0 && gy < HGT) {
                const float* p = img + ((b * 12 + 2 * ci2) * HGT + gy) * WID + x;
                unsigned int lo = f2bf(p[0]);
                unsigned int hi = f2bf(p[HGT * WID]);
                imgT32[(yi * IMG_W + x + 1) * 8 + ci2] = lo | (hi << 16);
            }
        }
    }
    __syncthreads();

    // ---- conv1 weight B fragments: K' = tap*16 + ci, 5 k-steps ----
    // B layout: lane holds N = lane%16, element e -> K' = 32*s + (e<8 ? e : e+8) + 8*half
    Frag bw1[5];
    {
        int n = (lm < 3) ? lm : 0;
        for (int s = 0; s < 5; ++s)
            for (int e = 0; e < 16; ++e) {
                int K   = 32 * s + (e < 8 ? e : e + 8) + 8 * half;
                int tap = K >> 4, ci = K & 15;
                bool ok = (lm < 3) && (tap < 9) && (ci < 12);
                float w = comp_w[n * 108 + (ok ? (ci * 9 + tap) : 0)];
                bw1[s].h[e] = ok ? f2bf(w) : (unsigned short)0;
            }
    }
    const float bias1 = (lm < 3) ? comp_b[(lm < 3) ? lm : 0] : 0.f;  // hoisted

    // ---- conv1 implicit GEMM: A chunks are contiguous b128 LDS loads ----
    // lane holds M = lane%16; chunk c of k-step s covers tap = 2s+c,
    // channels ci = 8*half .. 8*half+7 of pixel (rr+a, col+bb)
    for (int tIdx = wave; tIdx < C1_ROWS * 13; tIdx += 8) {
        int rr = tIdx / 13, t = tIdx % 13;
        if (2 * Y0 + rr >= HGT) continue;     // conv2's zero pad row (uniform)
        const int colBase = 16 * t + lm;
        v8f acc = vzero();
#pragma unroll
        for (int s = 0; s < 5; ++s) {
            Frag a;
            {
                const int tap = 2 * s, aa = tap / 3, bb = tap % 3;
                a.q[0] = *(const P128*)(imgT +
                          ((rr + aa) * IMG_W + colBase + bb) * 16 + 8 * half);
            }
            if (2 * s + 1 < 9) {
                const int tap = 2 * s + 1, aa = tap / 3, bb = tap % 3;
                a.q[1] = *(const P128*)(imgT +
                          ((rr + aa) * IMG_W + colBase + bb) * 16 + 8 * half);
            } else {
                P128 zz; zz.x = zz.y = zz.z = zz.w = 0u;
                a.q[1] = zz;
            }
            acc = wmma_bf16(a, bw1[s], acc);
        }
        // D: lane VGPR r -> pixel M = r + 8*half, channel N = lane%16.
        // Store unconditionally (cols 200..207 land in the pad region and
        // are re-zeroed below); pack two bf16 per b32 store.
        if (lm < 3) {
            unsigned int* dst32 = (unsigned int*)(c1T +
                lm * (C1_ROWS * C1_W) + rr * C1_W + 16 * t + 8 * half);
#pragma unroll
            for (int rp = 0; rp < 4; ++rp) {
                unsigned int lo = f2bf(acc[2 * rp]     + bias1);
                unsigned int hi = f2bf(acc[2 * rp + 1] + bias1);
                dst32[rp] = lo | (hi << 16);
            }
        }
    }
    __syncthreads();
    // re-zero conv2's zero-pad columns 200..207 (got garbage from tile t=12)
    if (tid < 3 * C1_ROWS * 8) {
        int ch = tid / (C1_ROWS * 8);
        int rem = tid - ch * (C1_ROWS * 8);
        int rrow = rem / 8, cc = 200 + (rem & 7);
        c1T[ch * (C1_ROWS * C1_W) + rrow * C1_W + cc] = 0;
    }
    __syncthreads();

    // ---- conv2 weights: K = c1*9 + tap (27 -> 32, 1 k-step), N = 32 ----
    Frag bw2a, bw2b;
    for (int e = 0; e < 16; ++e) {
        int K  = (e < 8 ? e : e + 8) + 8 * half;
        int Kc = (K < 27) ? K : 0;
        float wa = cnn_w[lm * 27 + Kc];
        float wb = cnn_w[(16 + lm) * 27 + Kc];
        bw2a.h[e] = (K < 27) ? f2bf(wa) : (unsigned short)0;
        bw2b.h[e] = (K < 27) ? f2bf(wb) : (unsigned short)0;
    }

    // ---- conv2 implicit GEMM + relu + spatial accumulation ----
    float acc0 = 0.f, acc1 = 0.f;
    const float cb0 = cnn_b[lm], cb1 = cnn_b[16 + lm];
    for (int mt = wave; mt < 2 * 7; mt += 8) {      // 2 rows x 7 col-tiles
        int rowIdx = mt / 7, t = mt % 7;
        Frag a;
        int X  = 16 * t + lm;
        int Xc = (X < 100) ? X : 0;
        for (int e = 0; e < 16; ++e) {
            int K  = (e < 8 ? e : e + 8) + 8 * half;
            int Kc = (K < 27) ? K : 0;
            unsigned short vv = c1T[k2Base[Kc] + rowIdx * (2 * C1_W) + 2 * Xc];
            a.h[e] = (K < 27 && X < 100) ? vv : (unsigned short)0;
        }
        v8f d0 = vzero(), d1 = vzero();
        d0 = wmma_bf16(a, bw2a, d0);
        d1 = wmma_bf16(a, bw2b, d1);
#pragma unroll
        for (int r = 0; r < 8; ++r) {
            int Xp = 16 * t + r + 8 * half;
            float msk = (Xp < 100) ? 1.f : 0.f;     // branchless (v_cndmask)
            acc0 += msk * fmaxf(d0[r] + cb0, 0.f);
            acc1 += msk * fmaxf(d1[r] + cb1, 0.f);
        }
    }
    atomicAdd(&accLds[lm],      acc0);
    atomicAdd(&accLds[16 + lm], acc1);
    __syncthreads();
    if (tid < 32) atomicAdd(&z_sum[b * 32 + tid], accLds[tid]);
}

// =================================================================
// Kernel 2: img_emb = (z_sum/10000) @ cnn_fc_w.T + b   (64 x 1000)
// =================================================================
__global__ __launch_bounds__(256) void k_fc(
    const float* __restrict__ z_sum, const float* __restrict__ fw,
    const float* __restrict__ fb, float* __restrict__ img_emb)
{
    int idx = blockIdx.x * 256 + threadIdx.x;
    if (idx >= 64 * 1000) return;
    int b = idx / 1000, o = idx % 1000;
    float s = fb[o];
    for (int c = 0; c < 32; ++c)
        s += (z_sum[b * 32 + c] * (1.0f / 10000.0f)) * fw[o * 32 + c];
    img_emb[idx] = s;
}

// =================================================================
// Kernel 3: pack [w_ih | w_hh] -> bf16, k-contiguous (512 x 192)
// =================================================================
__global__ __launch_bounds__(256) void k_prep(
    const float* __restrict__ w_ih, const float* __restrict__ w_hh,
    unsigned short* __restrict__ wcat)
{
    int idx = blockIdx.x * 256 + threadIdx.x;
    if (idx >= 512 * 192) return;
    int n = idx / 192, k = idx % 192;
    float v = (k < 64) ? w_ih[n * 64 + k] : w_hh[n * 128 + (k - 64)];
    wcat[idx] = f2bf(v);
}

// =================================================================
// Kernel 4: LSTM encode (9 steps) + autoregressive decode (10 steps)
// single workgroup, 8 waves; gates GEMM 64x512x192 via WMMA bf16
// =================================================================
__device__ __forceinline__ void lstm_gemm_cell(
    unsigned short* act, float* cst, float* gates,
    const unsigned short* __restrict__ wcat,
    const float* __restrict__ b_ih, const float* __restrict__ b_hh,
    int tid, int lane, int wave)
{
    const int half = lane >> 4, lm = lane & 15;
    // 4 M-tiles x 32 N-tiles over 8 waves, K = 6 steps of 32
    for (int pr = wave; pr < 128; pr += 8) {
        int mt = pr & 3, nt = pr >> 2;
        v8f acc = vzero();
#pragma unroll
        for (int s = 0; s < 6; ++s) {
            Frag a, bf;
            int arow = mt * 16 + lm;
            int akb  = 32 * s + 8 * half;                 // A: two 8-elem chunks
            a.q[0] = *(const P128*)(act + arow * 192 + akb);
            a.q[1] = *(const P128*)(act + arow * 192 + akb + 16);
            int n   = nt * 16 + lm;
            int bkb = 32 * s + 8 * half;                  // B: two 8-elem chunks
            const P128* bp = (const P128*)(wcat + n * 192 + bkb);
            bf.q[0] = bp[0];
            bf.q[1] = bp[2];
            acc = wmma_bf16(a, bf, acc);
        }
        for (int r = 0; r < 8; ++r) {
            int gm = mt * 16 + r + 8 * half;
            int gn = nt * 16 + lm;
            gates[gm * 512 + gn] = acc[r];
        }
    }
    __syncthreads();
    // cell nonlinearity: i,f,g,o = gates[:,0:128],[128:256],[256:384],[384:512]
    for (int i = tid; i < 64 * 128; i += 256) {
        int m = i >> 7, j = i & 127;
        const float* g = gates + m * 512;
        float gi = g[j]       + b_ih[j]       + b_hh[j];
        float gf = g[128 + j] + b_ih[128 + j] + b_hh[128 + j];
        float gg = g[256 + j] + b_ih[256 + j] + b_hh[256 + j];
        float go = g[384 + j] + b_ih[384 + j] + b_hh[384 + j];
        float c2 = sigm(gf) * cst[i] + sigm(gi) * tanhf(gg);
        cst[i] = c2;
        act[m * 192 + 64 + j] = f2bf(sigm(go) * tanhf(c2));
    }
    __syncthreads();
}

__global__ __launch_bounds__(256) void k_lstm(
    const float* __restrict__ hist_points, const float* __restrict__ hist_step,
    const float* __restrict__ embed_w, const float* __restrict__ embed_b,
    const float* __restrict__ h0, const float* __restrict__ c0,
    const float* __restrict__ b_ih, const float* __restrict__ b_hh,
    const float* __restrict__ out_w, const float* __restrict__ out_b,
    const float* __restrict__ img_emb, const unsigned short* __restrict__ wcat,
    float* __restrict__ gates, float* __restrict__ pred_traj)
{
    __shared__ unsigned short act[64 * 192];  // [embed(64) | h(128)] bf16
    __shared__ float cst[64 * 128];           // cell state f32
    __shared__ float pose[64 * 4];
    __shared__ float stepb[64 * 4];
    __shared__ float baseo[64 * 4];           // img_emb part of out layer

    const int tid = threadIdx.x, lane = tid & 31, wave = tid >> 5;

    // init h, c tiled from h0/c0
    for (int i = tid; i < 64 * 128; i += 256) {
        int j = i & 127;
        cst[i] = c0[j];
        act[(i >> 7) * 192 + 64 + j] = f2bf(h0[j]);
    }
    // pose = hist_points[:, 9]; baseo = img_emb @ out_w[:,128:].T + out_b
    {
        int m = tid >> 2, o = tid & 3;
        pose[tid] = hist_points[(m * 10 + 9) * 4 + o];
        float s = out_b[o];
        const float* wrow = out_w + o * 1128 + 128;
        const float* ie   = img_emb + m * 1000;
        for (int c2 = 0; c2 < 1000; ++c2) s += ie[c2] * wrow[c2];
        baseo[tid] = s;
    }
    __syncthreads();

    // ---- history encode: steps 1..9 of hist_points_step ----
    for (int t = 1; t <= 9; ++t) {
        for (int i = tid; i < 64 * 64; i += 256) {
            int m = i >> 6, j = i & 63;
            const float* sp = hist_step + (m * 10 + t) * 4;
            float s = embed_b[j];
            for (int q = 0; q < 4; ++q) s += sp[q] * embed_w[j * 4 + q];
            act[m * 192 + j] = f2bf(fmaxf(s, 0.f));
        }
        __syncthreads();
        lstm_gemm_cell(act, cst, gates, wcat, b_ih, b_hh, tid, lane, wave);
    }

    // ---- autoregressive decode: 10 steps ----
    for (int p = 0; p < 10; ++p) {
        {
            int m = tid >> 2, o = tid & 3;
            float s = baseo[tid];
            const float* wrow = out_w + o * 1128;
            for (int j = 0; j < 128; ++j) s += bf2f(act[m * 192 + 64 + j]) * wrow[j];
            float np = pose[tid] + s;
            pose[tid] = np;
            stepb[tid] = s;
            pred_traj[(m * 10 + p) * 4 + o] = np;
        }
        __syncthreads();
        for (int i = tid; i < 64 * 64; i += 256) {
            int m = i >> 6, j = i & 63;
            float s = embed_b[j];
            for (int q = 0; q < 4; ++q) s += stepb[m * 4 + q] * embed_w[j * 4 + q];
            act[m * 192 + j] = f2bf(fmaxf(s, 0.f));
        }
        __syncthreads();
        lstm_gemm_cell(act, cst, gates, wcat, b_ih, b_hh, tid, lane, wave);
    }
}

// =================================================================
// Kernel 5: Gaussian rasterizer. grid = 640 (b*10+t), block = 256
// ga = gc = 1/(2*1.8^2), gb = 0 (THETA = 0)
// =================================================================
__global__ __launch_bounds__(256) void k_raster(
    const float* __restrict__ traj, const float* __restrict__ ego_in,
    float* __restrict__ outp)
{
    const int bt = blockIdx.x;
    const int b  = bt / 10;
    const float x  = traj[bt * 4 + 0];
    const float y  = traj[bt * 4 + 1];
    const float hd = traj[bt * 4 + 2];
    const float ego = ego_in[b];
    const float ho  = -(ego - 1.57079632679489662f);
    const float cho = cosf(ho), sho = sinf(ho);
    const float px = 100.0f + (x * cho - y * sho) * 10.0f;
    const float py = 160.0f - (x * sho + y * cho) * 10.0f;
    const float bh = -hd - ego;
    const float cbh = cosf(bh), sbh = sinf(bh);
    const float ga = 0.154320987654320987f;      // 1/(2*1.8^2)
    const float s1 = 14.235f, s2 = 28.47f;       // CENTER/FRONT shift / RES
    const float cx0 = px,             cy0 = py;
    const float cx1 = px + s1 * cbh,  cy1 = py - s1 * sbh;
    const float cx2 = px + s2 * cbh,  cy2 = py - s2 * sbh;
    float* dst = outp + 2560 + bt * 40000;
    for (int idx = threadIdx.x; idx < 40000; idx += 256) {
        float p = (float)(idx / 200);   // first grid axis (gx = i)
        float q = (float)(idx % 200);   // second grid axis (gy = j)
        float d0x = p - cx0, d0y = q - cy0;
        float d1x = p - cx1, d1y = q - cy1;
        float d2x = p - cx2, d2y = q - cy2;
        float v0 = __expf(-ga * (d0x * d0x + d0y * d0y));
        float v1 = __expf(-ga * (d1x * d1x + d1y * d1y));
        float v2 = __expf(-ga * (d2x * d2x + d2y * d2y));
        dst[idx] = fmaxf(v0, fmaxf(v1, v2));
    }
}

// =================================================================
extern "C" void kernel_launch(void* const* d_in, const int* in_sizes, int n_in,
                              void* d_out, int out_size, void* d_ws, size_t ws_size,
                              hipStream_t stream)
{
    (void)in_sizes; (void)n_in; (void)out_size; (void)ws_size;
    const float* img    = (const float*)d_in[0];
    const float* hist   = (const float*)d_in[1];
    const float* histst = (const float*)d_in[2];
    const float* ego    = (const float*)d_in[3];
    const float* comp_w = (const float*)d_in[4];
    const float* comp_b = (const float*)d_in[5];
    const float* cnn_w  = (const float*)d_in[6];
    const float* cnn_b  = (const float*)d_in[7];
    const float* fc_w   = (const float*)d_in[8];
    const float* fc_b   = (const float*)d_in[9];
    const float* emb_w  = (const float*)d_in[10];
    const float* emb_b  = (const float*)d_in[11];
    const float* h0     = (const float*)d_in[12];
    const float* c0     = (const float*)d_in[13];
    const float* w_ih   = (const float*)d_in[14];
    const float* w_hh   = (const float*)d_in[15];
    const float* b_ih   = (const float*)d_in[16];
    const float* b_hh   = (const float*)d_in[17];
    const float* out_w  = (const float*)d_in[18];
    const float* out_b  = (const float*)d_in[19];
    float* outp = (float*)d_out;

    // workspace layout (all offsets 256B aligned)
    char* ws = (char*)d_ws;
    float*          z_sum   = (float*)(ws + 0);                 //   8,192 B
    float*          img_emb = (float*)(ws + 8192);              // 256,000 B
    float*          gates   = (float*)(ws + 8192 + 256000);     // 131,072 B
    unsigned short* wcat    = (unsigned short*)(ws + 8192 + 256000 + 131072);

    hipMemsetAsync(z_sum, 0, 64 * 32 * sizeof(float), stream);
    k_conv<<<dim3(64, 50), 256, 0, stream>>>(img, comp_w, comp_b, cnn_w, cnn_b, z_sum);
    k_fc  <<<250, 256, 0, stream>>>(z_sum, fc_w, fc_b, img_emb);
    k_prep<<<384, 256, 0, stream>>>(w_ih, w_hh, wcat);
    k_lstm<<<1, 256, 0, stream>>>(hist, histst, emb_w, emb_b, h0, c0,
                                  b_ih, b_hh, out_w, out_b, img_emb, wcat,
                                  gates, outp);
    k_raster<<<640, 256, 0, stream>>>(outp, ego, outp);
}
